// Net_66468913873438
// MI455X (gfx1250) — matrix operand order repaired and checked by reading, not compiled, not run on hardware
//
#include <hip/hip_runtime.h>
#include <cstdint>
#include <cstddef>

// ---------------------------------------------------------------------------
// CDNA5 (gfx1250) types for WMMA: wave32, 16x16x32 bf16 -> f32 accumulate.
// ---------------------------------------------------------------------------
typedef __attribute__((ext_vector_type(16))) __bf16 v16bf;
typedef __attribute__((ext_vector_type(8)))  float  v8f;

union Frag16 { v16bf v; unsigned int u[8]; };

__device__ __forceinline__ unsigned short f2bf(float f) {
  unsigned int u = __float_as_uint(f);
  u += 0x7FFFu + ((u >> 16) & 1u);      // round-to-nearest-even
  return (unsigned short)(u >> 16);
}

// ---------------------------------------------------------------------------
// Small utility kernels
// ---------------------------------------------------------------------------
__global__ void k_fill(float* __restrict__ p, float v, int n) {
  int i = blockIdx.x * blockDim.x + threadIdx.x;
  if (i < n) p[i] = v;
}

__global__ void k_cvt_bf16(const float* __restrict__ s, unsigned short* __restrict__ d, int n) {
  int i = blockIdx.x * blockDim.x + threadIdx.x;
  if (i < n) d[i] = f2bf(s[i]);
}

// src is [R][C] row-major; dst is [C][R] row-major (i.e. transpose)
__global__ void k_cvt_bf16_t(const float* __restrict__ s, unsigned short* __restrict__ d,
                             int R, int C) {
  int i = blockIdx.x * blockDim.x + threadIdx.x;
  if (i < R * C) {
    int r = i / C, c = i % C;
    d[(size_t)c * R + r] = f2bf(s[i]);
  }
}

__global__ void k_deg_scatter(const int* __restrict__ col, const float* __restrict__ w,
                              float* __restrict__ deg, int E) {
  int e = blockIdx.x * blockDim.x + threadIdx.x;
  if (e < E) atomicAdd(&deg[col[e]], w[e]);
}

__global__ void k_dinv(const float* __restrict__ deg, float* __restrict__ dinv, int n) {
  int i = blockIdx.x * blockDim.x + threadIdx.x;
  if (i < n) {
    float d = deg[i];
    dinv[i] = d > 0.f ? rsqrtf(d) : 0.f;
  }
}

// ---------------------------------------------------------------------------
// WMMA GEMM: C[M x Nc] = A[M x K] (fp32) * B[K x Nc] (bf16) (+ bias[col])
// 256 threads = 8 wave32s. Block tile MT x NT, K-step 32.
//   NT=64 -> MT=32 (2 row-halves x 4 col-tiles)
//   NT=16 -> MT=128 (8 row-tiles x 1 col-tile)
// A staged to LDS with fp32->bf16 conversion; B staged transposed so both
// fragments are built from dword-aligned ds loads.
// ---------------------------------------------------------------------------
template <int NT, int MT>
__global__ __launch_bounds__(256) void k_gemm_wmma(
    const float* __restrict__ A, const unsigned short* __restrict__ B,
    float* __restrict__ C, int M, int K, int Nc, const float* __restrict__ bias) {
  static_assert(NT == 16 || NT == 64, "NT must be 16 or 64");
  constexpr int KT = 32;
  constexpr int KTP = 40;                 // padded LDS stride (elements)
  constexpr int NWT = NT / 16;
  static_assert(MT == 16 * (8 / NWT), "MT must match 8 waves");

  __shared__ unsigned short sA[MT][KTP];
  __shared__ unsigned short sBT[NT][KTP];

  const int tid   = threadIdx.x;
  const int lane  = tid & 31;
  const int wave  = tid >> 5;
  const int waveN = wave % NWT;
  const int waveM = wave / NWT;
  const int rowBase = blockIdx.x * MT;
  const int colBase = blockIdx.y * NT;
  const int half = lane >> 4;
  const int l16  = lane & 15;

  v8f acc = {};

  for (int k0 = 0; k0 < K; k0 += KT) {
    // ---- stage A tile (MT x 32 fp32 -> bf16), float4 coalesced loads ----
    for (int i = tid * 4; i < MT * KT; i += 256 * 4) {
      int r = i / KT, kk = i % KT;
      int gr = rowBase + r;
      float4 av;
      if (gr < M) av = *(const float4*)(A + (size_t)gr * K + (k0 + kk));
      else        av = make_float4(0.f, 0.f, 0.f, 0.f);
      unsigned int p0 = (unsigned int)f2bf(av.x) | ((unsigned int)f2bf(av.y) << 16);
      unsigned int p1 = (unsigned int)f2bf(av.z) | ((unsigned int)f2bf(av.w) << 16);
      *(unsigned int*)&sA[r][kk]     = p0;
      *(unsigned int*)&sA[r][kk + 2] = p1;
    }
    // ---- stage B tile transposed (already bf16 in global) ----
    for (int i = tid; i < KT * NT; i += 256) {
      int kk = i / NT, n = i % NT;
      sBT[n][kk] = B[(size_t)(k0 + kk) * Nc + (colBase + n)];
    }
    // prefetch next A K-slab (global_prefetch_b8)
    if (k0 + KT < K) {
      int pr = rowBase + (tid >> 3);
      if (pr >= M) pr = M - 1;
      __builtin_prefetch(A + (size_t)pr * K + (k0 + KT) + ((tid & 7) << 2), 0, 0);
    }
    __syncthreads();

    // ---- build fragments per ISA 7.12.2 layouts ----
    Frag16 afrag, bfrag;
    const int arow = waveM * 16 + l16;
    const int bcol = waveN * 16 + l16;
#pragma unroll
    for (int p = 0; p < 8; ++p) {
      // A: k(s) = ((s&8)<<1) + half*8 + (s&7), s = 2p (pairs adjacent)
      const int ka = ((p & 4) << 2) + (half << 3) + ((p & 3) << 1);
      afrag.u[p] = *(const unsigned int*)&sA[arow][ka];
      // B: k(s) = ((s&8)<<1) + half*8 + (s&7), same as A
      const int kb = ((p & 4) << 2) + (half << 3) + ((p & 3) << 1);
      bfrag.u[p] = *(const unsigned int*)&sBT[bcol][kb];
    }
    acc = __builtin_amdgcn_wmma_f32_16x16x32_bf16(
        /*neg_a=*/false, afrag.v, /*neg_b=*/false, bfrag.v,
        /*c_mod=*/(short)0, acc, /*reuse_a=*/false, /*reuse_b=*/false);
    __syncthreads();
  }

  // ---- epilogue: D layout -> m = half*8 + j, n = lane&15 ----
  const int gc = colBase + waveN * 16 + l16;
  const float badd = bias ? bias[gc] : 0.f;
#pragma unroll
  for (int j = 0; j < 8; ++j) {
    const int m  = half * 8 + j;
    const int gr = rowBase + waveM * 16 + m;
    if (gr < M) C[(size_t)gr * Nc + gc] = acc[j] + badd;
  }
}

// ---------------------------------------------------------------------------
// GCN self-loop init: out[i] = dinv[i]^2 * h[i]   (float4 per thread)
// ---------------------------------------------------------------------------
template <int F>
__global__ void k_self_init(const float* __restrict__ h, const float* __restrict__ dinv,
                            float* __restrict__ out, int n) {
  constexpr int Q = F / 4;
  int idx = blockIdx.x * blockDim.x + threadIdx.x;
  if (idx >= n * Q) return;
  int i = idx / Q;
  int f = (idx % Q) * 4;
  float di = dinv[i];
  float s = di * di;
  float4 hv = *(const float4*)(h + (size_t)i * F + f);
  float4 ov = make_float4(s * hv.x, s * hv.y, s * hv.z, s * hv.w);
  *(float4*)(out + (size_t)i * F + f) = ov;
}

// ---------------------------------------------------------------------------
// Edge scatter-add: out[col] += coef * h[row], coef = w (* dinv[r]*dinv[c])
// 4 features per thread -> fp32 global atomics, L2-resident accumulators.
// ---------------------------------------------------------------------------
template <int F, bool NORM>
__global__ void k_edge_scatter(const float* __restrict__ h, const int* __restrict__ row,
                               const int* __restrict__ col, const float* __restrict__ w,
                               const float* __restrict__ dinv, float* __restrict__ out,
                               int E) {
  constexpr int Q = F / 4;
  int gid = blockIdx.x * blockDim.x + threadIdx.x;
  int e = gid / Q;
  if (e >= E) return;
  int f = (gid % Q) * 4;
  int r = row[e], c = col[e];
  float coef = w[e];
  if constexpr (NORM) coef *= dinv[r] * dinv[c];
  float4 hv = *(const float4*)(h + (size_t)r * F + f);
  float* o = out + (size_t)c * F + f;
  atomicAdd(o + 0, coef * hv.x);
  atomicAdd(o + 1, coef * hv.y);
  atomicAdd(o + 2, coef * hv.z);
  atomicAdd(o + 3, coef * hv.w);
}

__global__ void k_bias_relu(const float* __restrict__ agg, const float* __restrict__ b,
                            float* __restrict__ x, int n64) {
  int idx = blockIdx.x * blockDim.x + threadIdx.x;
  if (idx < n64) x[idx] = fmaxf(agg[idx] + b[idx & 63], 0.f);
}

// GRU cell combine (H = 64, gates packed [r|z|n] along 3H)
__global__ void k_gru(const float* __restrict__ gi, const float* __restrict__ gh,
                      float* __restrict__ x, int n) {
  int idx = blockIdx.x * blockDim.x + threadIdx.x;
  if (idx >= n * 64) return;
  int i = idx >> 6, f = idx & 63;
  const float* pi = gi + (size_t)i * 192;
  const float* ph = gh + (size_t)i * 192;
  float r  = 1.f / (1.f + __expf(-(pi[f]      + ph[f])));
  float z  = 1.f / (1.f + __expf(-(pi[64 + f] + ph[64 + f])));
  float ng = tanhf(pi[128 + f] + r * ph[128 + f]);
  float xo = x[idx];
  x[idx] = (1.f - z) * ng + z * xo;
}

// out = log_softmax(agg + b2) over C=16, one node per thread
__global__ void k_logsoftmax(const float* __restrict__ agg, const float* __restrict__ b,
                             float* __restrict__ out, int n) {
  int i = blockIdx.x * blockDim.x + threadIdx.x;
  if (i >= n) return;
  float v[16];
  float mx = -3.402823466e38f;
#pragma unroll
  for (int j = 0; j < 16; ++j) {
    v[j] = agg[(size_t)i * 16 + j] + b[j];
    mx = fmaxf(mx, v[j]);
  }
  float s = 0.f;
#pragma unroll
  for (int j = 0; j < 16; ++j) s += __expf(v[j] - mx);
  float lse = __logf(s) + mx;
#pragma unroll
  for (int j = 0; j < 16; ++j) out[(size_t)i * 16 + j] = v[j] - lse;
}

// ---------------------------------------------------------------------------
// Host orchestration
// ---------------------------------------------------------------------------
extern "C" void kernel_launch(void* const* d_in, const int* in_sizes, int n_in,
                              void* d_out, int out_size, void* d_ws, size_t ws_size,
                              hipStream_t stream) {
  (void)n_in; (void)out_size; (void)ws_size;
  const float* x   = (const float*)d_in[0];
  const int*   ei  = (const int*)d_in[1];
  const float* ew  = (const float*)d_in[2];
  const float* W1  = (const float*)d_in[3];
  const float* b1  = (const float*)d_in[4];
  const float* Wg  = (const float*)d_in[5];
  const float* Wih = (const float*)d_in[6];
  const float* Whh = (const float*)d_in[7];
  const float* bih = (const float*)d_in[8];
  const float* bhh = (const float*)d_in[9];
  const float* W2  = (const float*)d_in[10];
  const float* b2  = (const float*)d_in[11];
  float* out = (float*)d_out;

  const int FIN = 512, H = 64, C = 16;
  const int N = in_sizes[0] / FIN;
  const int E = in_sizes[2];
  const int* row = ei;       // edge_index[0] = source (gather)
  const int* col = ei + E;   // edge_index[1] = destination (scatter)

  // ---- workspace layout (256B-aligned) ----
  char* ws = (char*)d_ws;
  size_t off = 0;
  auto alloc = [&](size_t bytes) -> void* {
    void* p = ws + off;
    off = (off + bytes + 255) & ~((size_t)255);
    return p;
  };
  float* deg   = (float*)alloc((size_t)N * 4);
  float* dinv  = (float*)alloc((size_t)N * 4);
  float* bufH  = (float*)alloc((size_t)N * H * 4);       // GEMM outputs (h1 / m / h2)
  float* bufA  = (float*)alloc((size_t)N * H * 4);       // aggregation accumulator
  float* bufX  = (float*)alloc((size_t)N * H * 4);       // current node features
  float* bufGi = (float*)alloc((size_t)N * 3 * H * 4);
  float* bufGh = (float*)alloc((size_t)N * 3 * H * 4);
  unsigned short* W1bf  = (unsigned short*)alloc((size_t)FIN * H * 2);
  unsigned short* Wg0bf = (unsigned short*)alloc((size_t)H * H * 2);
  unsigned short* Wg1bf = (unsigned short*)alloc((size_t)H * H * 2);
  unsigned short* WihT  = (unsigned short*)alloc((size_t)H * 3 * H * 2);
  unsigned short* WhhT  = (unsigned short*)alloc((size_t)H * 3 * H * 2);
  unsigned short* W2bf  = (unsigned short*)alloc((size_t)H * C * 2);

  const int B256 = 256;
  auto cdiv = [](long long a, long long b) { return (int)((a + b - 1) / b); };

  // ---- weight prep (bf16, transpose for @W.T) ----
  k_cvt_bf16  <<<cdiv(FIN * H, B256), B256, 0, stream>>>(W1, W1bf, FIN * H);
  k_cvt_bf16  <<<cdiv(H * H, B256),   B256, 0, stream>>>(Wg, Wg0bf, H * H);
  k_cvt_bf16  <<<cdiv(H * H, B256),   B256, 0, stream>>>(Wg + H * H, Wg1bf, H * H);
  k_cvt_bf16_t<<<cdiv(3 * H * H, B256), B256, 0, stream>>>(Wih, WihT, 3 * H, H);
  k_cvt_bf16_t<<<cdiv(3 * H * H, B256), B256, 0, stream>>>(Whh, WhhT, 3 * H, H);
  k_cvt_bf16  <<<cdiv(H * C, B256),   B256, 0, stream>>>(W2, W2bf, H * C);

  // ---- GCN normalization (shared by both GCN convs) ----
  k_fill       <<<cdiv(N, B256), B256, 0, stream>>>(deg, 1.0f, N);   // self loops
  k_deg_scatter<<<cdiv(E, B256), B256, 0, stream>>>(col, ew, deg, E);
  k_dinv       <<<cdiv(N, B256), B256, 0, stream>>>(deg, dinv, N);

  // ---- GCN conv 1: h = x @ W1 ; agg ; relu(agg + b1) ----
  dim3 g64(cdiv(N, 32), 1);
  k_gemm_wmma<64, 32><<<g64, B256, 0, stream>>>(x, W1bf, bufH, N, FIN, H, nullptr);
  k_self_init<64><<<cdiv((long long)N * 16, B256), B256, 0, stream>>>(bufH, dinv, bufA, N);
  k_edge_scatter<64, true><<<cdiv((long long)E * 16, B256), B256, 0, stream>>>(
      bufH, row, col, ew, dinv, bufA, E);
  k_bias_relu<<<cdiv((long long)N * H, B256), B256, 0, stream>>>(bufA, b1, bufX, N * H);

  // ---- GatedGraphConv: 2 layers ----
  const unsigned short* WgL[2] = {Wg0bf, Wg1bf};
  dim3 g192(cdiv(N, 32), 3);
  for (int L = 0; L < 2; ++L) {
    k_gemm_wmma<64, 32><<<g64, B256, 0, stream>>>(bufX, WgL[L], bufH, N, H, H, nullptr);
    k_fill<<<cdiv((long long)N * H, B256), B256, 0, stream>>>(bufA, 0.f, N * H);
    k_edge_scatter<64, false><<<cdiv((long long)E * 16, B256), B256, 0, stream>>>(
        bufH, row, col, ew, nullptr, bufA, E);
    k_gemm_wmma<64, 32><<<g192, B256, 0, stream>>>(bufA, WihT, bufGi, N, H, 3 * H, bih);
    k_gemm_wmma<64, 32><<<g192, B256, 0, stream>>>(bufX, WhhT, bufGh, N, H, 3 * H, bhh);
    k_gru<<<cdiv((long long)N * H, B256), B256, 0, stream>>>(bufGi, bufGh, bufX, N);
  }

  // ---- GCN conv 2 (C=16) + log_softmax ----
  dim3 g16(cdiv(N, 128), 1);
  k_gemm_wmma<16, 128><<<g16, B256, 0, stream>>>(bufX, W2bf, bufH, N, H, C, nullptr);
  k_self_init<16><<<cdiv((long long)N * 4, B256), B256, 0, stream>>>(bufH, dinv, bufA, N);
  k_edge_scatter<16, true><<<cdiv((long long)E * 4, B256), B256, 0, stream>>>(
      bufH, row, col, ew, dinv, bufA, E);
  k_logsoftmax<<<cdiv(N, B256), B256, 0, stream>>>(bufA, b2, out, N);
}
